// KRNN7_14035953124004
// MI455X (gfx1250) — matrix operand order repaired and checked by reading, not compiled, not run on hardware
//
#include <hip/hip_runtime.h>
#include <hip/hip_bf16.h>

typedef __attribute__((ext_vector_type(16))) __bf16 v16bf;
typedef __attribute__((ext_vector_type(8)))  float  v8f;
typedef __attribute__((ext_vector_type(4)))  int    i4;

#define BN_TOT 10400
#define NNODE  325
#define NCLUST 10
#define HID    64
#define G3     192
#define TSTEPS 12
#define TOUT   12
#define ROWS_PER_BLOCK 128
#define WAVES  8

struct alignas(16) SharedMem {
    unsigned       bpack[12][2][32][8];              // 24576 B  (B fragments, bf16x2)
    unsigned short hstage[WAVES][16][64];            // 16384 B  (h as bf16, C->A relayout)
    float          xs[ROWS_PER_BLOCK][TSTEPS][2];    // 12288 B
    float          ewih0[G3], ewih1[G3];             // 1536 B
    float          ebih[G3], ebhh[G3];               // 1536 B
    float          dwih[G3];                         // 768 B
    float          dbih[G3], dbhh[G3];               // 1536 B
    float          linw[HID];                        // 256 B
    float          lv[ROWS_PER_BLOCK];               // 512 B
    float          wsm[ROWS_PER_BLOCK];              // 512 B
    float          linb;
};                                                   // ~59.9 KB total

union BFrag { unsigned u[8]; i4 q[2]; v16bf v; };

__device__ __forceinline__ unsigned short f2bf(float f) {
    return (unsigned short)((__float_as_uint(f) + 0x8000u) >> 16);
}
__device__ __forceinline__ float fast_rcp(float x) {
    return __builtin_amdgcn_rcpf(x);
}
__device__ __forceinline__ float sigmoidf(float x) {
    return fast_rcp(1.0f + __expf(-x));      // v_exp + v_rcp
}
__device__ __forceinline__ float tanh_fast(float x) {
    float e = __expf(2.0f * x);
    return (e - 1.0f) * fast_rcp(e + 1.0f);
}

// Pre-swizzle a (192 x 64) f32 weight matrix into bf16 B-fragment order:
// bpack[tile][kc][lane][v2] = {B[K][n], B[K+1][n]}, B[K][g] = W[g*64+K],
// K = 2*(v2&3) + 8*(lane>>4) + 16*(v2>>2) + 32*kc, n = lane&15, g = 16*tile + n.
__device__ __forceinline__ void pack_B(SharedMem& sm, const float* __restrict__ W) {
    for (int idx = threadIdx.x; idx < 12 * 2 * 32 * 8; idx += blockDim.x) {
        int v2   = idx & 7;
        int lane = (idx >> 3) & 31;
        int kc   = (idx >> 8) & 1;
        int tile = idx >> 9;
        int K = 2 * (v2 & 3) + 8 * (lane >> 4) + 16 * (v2 >> 2) + 32 * kc;
        int n = lane & 15;
        int g = tile * 16 + n;
        float f0 = W[g * 64 + K];
        float f1 = W[g * 64 + K + 1];
        sm.bpack[tile][kc][lane][v2] =
            (unsigned)f2bf(f0) | ((unsigned)f2bf(f1) << 16);
    }
}

// Pull the whole B-pack into registers (loop-invariant across 12 timesteps).
// Each lane's fragment is 32 contiguous bytes -> 2x ds_load_b128.
__device__ __forceinline__ void load_bfrags(SharedMem& sm, BFrag bfr[12][2], int lane) {
#pragma unroll
    for (int tile = 0; tile < 12; ++tile)
#pragma unroll
        for (int kc = 0; kc < 2; ++kc) {
            const i4* p = (const i4*)&sm.bpack[tile][kc][lane][0];
            bfr[tile][kc].q[0] = p[0];
            bfr[tile][kc].q[1] = p[1];
        }
}

// One GRU step for this wave's 16 rows. h kept in C/D-fragment layout:
// hfrag[t4][v] = h(m = v + 8*(lane/16), j = 16*t4 + (lane&15)).
template <bool ENC>
__device__ __forceinline__ void gru_step(SharedMem& sm, float hfrag[4][8],
                                         BFrag bfr[12][2],
                                         int wave, int lane, int step) {
    const int half = lane >> 4;
    const int nl   = lane & 15;

    // ---- stage h to LDS as bf16, wave-private tile ----
    unsigned short* hs = &sm.hstage[wave][0][0];
#pragma unroll
    for (int t4 = 0; t4 < 4; ++t4)
#pragma unroll
        for (int v = 0; v < 8; ++v)
            hs[(v + 8 * half) * 64 + t4 * 16 + nl] = f2bf(hfrag[t4][v]);
    asm volatile("s_wait_dscnt 0" ::: "memory");

    // ---- reload as A fragments (16x32 bf16 layout): 4x ds_load_b128 ----
    // A vgpr quads are contiguous: K = {8h..8h+7}, {16+8h..}, {32+8h..}, {48+8h..}
    BFrag a0, a1;
    const unsigned short* hrow = hs + nl * 64 + 8 * half;
    a0.q[0] = *(const i4*)(hrow +  0);
    a0.q[1] = *(const i4*)(hrow + 16);
    a1.q[0] = *(const i4*)(hrow + 32);
    a1.q[1] = *(const i4*)(hrow + 48);

    // ---- hoist per-row inputs (x_t or last value) ----
    float x0v[8], x1v[8];
#pragma unroll
    for (int v = 0; v < 8; ++v) {
        int lr = wave * 16 + v + 8 * half;
        if (ENC) {
            x0v[v] = sm.xs[lr][step][0];
            x1v[v] = sm.xs[lr][step][1];
        } else {
            x0v[v] = sm.lv[lr];
            x1v[v] = 0.f;
        }
    }

    // ---- init accumulators ----
    v8f acc[12];
#pragma unroll
    for (int tile = 0; tile < 8; ++tile) {       // r,z tiles: gi + bih + bhh
        int g = tile * 16 + nl;
        float bsum = ENC ? (sm.ebih[g] + sm.ebhh[g]) : (sm.dbih[g] + sm.dbhh[g]);
        float w0   = ENC ? sm.ewih0[g] : sm.dwih[g];
        float w1   = ENC ? sm.ewih1[g] : 0.f;
#pragma unroll
        for (int v = 0; v < 8; ++v) {
            float gi = fmaf(x0v[v], w0, bsum);
            if (ENC) gi = fmaf(x1v[v], w1, gi);
            acc[tile][v] = gi;
        }
    }
#pragma unroll
    for (int tile = 8; tile < 12; ++tile) {      // n tiles: bhh only
        int g = tile * 16 + nl;
        float b = ENC ? sm.ebhh[g] : sm.dbhh[g];
#pragma unroll
        for (int v = 0; v < 8; ++v) acc[tile][v] = b;
    }

    // ---- gh GEMM: K-chunk-grouped so dependent WMMAs on one acc are 12 apart ----
#pragma unroll
    for (int tile = 0; tile < 12; ++tile)
        acc[tile] = __builtin_amdgcn_wmma_f32_16x16x32_bf16(
            false, a0.v, false, bfr[tile][0].v, (short)0, acc[tile], false, false);
#pragma unroll
    for (int tile = 0; tile < 12; ++tile)
        acc[tile] = __builtin_amdgcn_wmma_f32_16x16x32_bf16(
            false, a1.v, false, bfr[tile][1].v, (short)0, acc[tile], false, false);

    // ---- GRU gate update (r,z,n share lane/vgpr position across tile groups) ----
#pragma unroll
    for (int t4 = 0; t4 < 4; ++t4) {
        int gn = 128 + t4 * 16 + nl;
        float bihn = ENC ? sm.ebih[gn] : sm.dbih[gn];
        float wn0  = ENC ? sm.ewih0[gn] : sm.dwih[gn];
        float wn1  = ENC ? sm.ewih1[gn] : 0.f;
#pragma unroll
        for (int v = 0; v < 8; ++v) {
            float r = sigmoidf(acc[t4][v]);
            float z = sigmoidf(acc[4 + t4][v]);
            float gin = fmaf(x0v[v], wn0, bihn);
            if (ENC) gin = fmaf(x1v[v], wn1, gin);
            float nn = tanh_fast(fmaf(r, acc[8 + t4][v], gin));
            hfrag[t4][v] = fmaf(z, hfrag[t4][v] - nn, nn);  // n + z*(h-n)
        }
    }
}

__global__ void __launch_bounds__(256)
krnn_zero(float* __restrict__ out, int n) {
    int i = blockIdx.x * blockDim.x + threadIdx.x;
    if (i < n) out[i] = 0.f;
}

__global__ void __launch_bounds__(256)
krnn_gru(const float* __restrict__ X,    const float* __restrict__ eWih,
         const float* __restrict__ eWhh, const float* __restrict__ eBih,
         const float* __restrict__ eBhh, const float* __restrict__ dWih,
         const float* __restrict__ dWhh, const float* __restrict__ dBih,
         const float* __restrict__ dBhh, const float* __restrict__ linW,
         const float* __restrict__ linB, const float* __restrict__ embed,
         float* __restrict__ out) {
    __shared__ SharedMem sm;
    const int c    = blockIdx.y;
    const int row0 = blockIdx.x * ROWS_PER_BLOCK;
    const int tid  = threadIdx.x;

    // ================= prepack =================
    for (int i = tid; i < G3; i += blockDim.x) {
        sm.ewih0[i] = eWih[c * G3 * 2 + i * 2 + 0];
        sm.ewih1[i] = eWih[c * G3 * 2 + i * 2 + 1];
        sm.ebih[i]  = eBih[c * G3 + i];
        sm.ebhh[i]  = eBhh[c * G3 + i];
        sm.dwih[i]  = dWih[c * G3 + i];
        sm.dbih[i]  = dBih[c * G3 + i];
        sm.dbhh[i]  = dBhh[c * G3 + i];
    }
    if (tid < HID) sm.linw[tid] = linW[c * HID + tid];
    if (tid == 0)  sm.linb = linB[c];
    if (tid < ROWS_PER_BLOCK) {
        int row = row0 + tid;
        if (row < BN_TOT) {
#pragma unroll
            for (int t = 0; t < TSTEPS; ++t) {
                sm.xs[tid][t][0] = X[row * TSTEPS * 2 + t * 2 + 0];
                sm.xs[tid][t][1] = X[row * TSTEPS * 2 + t * 2 + 1];
            }
            int n = row % NNODE;
            float mx = -1e30f;
            for (int cc = 0; cc < NCLUST; ++cc)
                mx = fmaxf(mx, embed[n * NCLUST + cc]);
            float s = 0.f;
            for (int cc = 0; cc < NCLUST; ++cc)
                s += __expf(embed[n * NCLUST + cc] - mx);
            sm.wsm[tid] = __expf(embed[n * NCLUST + c] - mx) * fast_rcp(s);
            sm.lv[tid]  = sm.xs[tid][TSTEPS - 1][0];
        } else {
#pragma unroll
            for (int t = 0; t < TSTEPS; ++t) {
                sm.xs[tid][t][0] = 0.f; sm.xs[tid][t][1] = 0.f;
            }
            sm.wsm[tid] = 0.f;
            sm.lv[tid]  = 0.f;
        }
    }
    pack_B(sm, eWhh + c * G3 * HID);
    __syncthreads();

    const int wave = tid >> 5;
    const int lane = tid & 31;
    const int half = lane >> 4;
    const int nl   = lane & 15;

    float hfrag[4][8];
#pragma unroll
    for (int t4 = 0; t4 < 4; ++t4)
#pragma unroll
        for (int v = 0; v < 8; ++v) hfrag[t4][v] = 0.f;

    BFrag bfr[12][2];

    // ================= encoder (12 steps) =================
    load_bfrags(sm, bfr, lane);
    for (int step = 0; step < TSTEPS; ++step)
        gru_step<true>(sm, hfrag, bfr, wave, lane, step);

    // swap resident weight pack to decoder Whh
    __syncthreads();
    pack_B(sm, dWhh + c * G3 * HID);
    __syncthreads();
    load_bfrags(sm, bfr, lane);

    // ================= decoder (12 steps) =================
    for (int step = 0; step < TOUT; ++step) {
        gru_step<false>(sm, hfrag, bfr, wave, lane, step);

        // value(m) = sum_j h(m,j)*linw[j] + linb ; reduce across 16 lanes / half
        float p[8];
#pragma unroll
        for (int v = 0; v < 8; ++v) {
            p[v] = hfrag[0][v] * sm.linw[nl]
                 + hfrag[1][v] * sm.linw[16 + nl]
                 + hfrag[2][v] * sm.linw[32 + nl]
                 + hfrag[3][v] * sm.linw[48 + nl];
        }
#pragma unroll
        for (int v = 0; v < 8; ++v) {
            p[v] += __shfl_xor(p[v], 1, 32);
            p[v] += __shfl_xor(p[v], 2, 32);
            p[v] += __shfl_xor(p[v], 4, 32);
            p[v] += __shfl_xor(p[v], 8, 32);
        }
        if (nl == 0) {
#pragma unroll
            for (int v = 0; v < 8; ++v) {
                int lr = wave * 16 + v + 8 * half;
                float val = p[v] + sm.linb;
                sm.lv[lr] = val;  // next step's decoder input (wave-private rows)
                int row = row0 + lr;
                if (row < BN_TOT)
                    atomicAdd(&out[row * TOUT + step], sm.wsm[lr] * val);
            }
        }
        asm volatile("s_wait_dscnt 0" ::: "memory");
    }
}

extern "C" void kernel_launch(void* const* d_in, const int* in_sizes, int n_in,
                              void* d_out, int out_size, void* d_ws, size_t ws_size,
                              hipStream_t stream) {
    // setup_inputs order: A(0, unused), X(1), enc_Wih(2), enc_Whh(3), enc_bih(4),
    // enc_bhh(5), dec_Wih(6), dec_Whh(7), dec_bih(8), dec_bhh(9), lin_W(10),
    // lin_b(11), embed(12)
    const float* X     = (const float*)d_in[1];
    const float* eWih  = (const float*)d_in[2];
    const float* eWhh  = (const float*)d_in[3];
    const float* eBih  = (const float*)d_in[4];
    const float* eBhh  = (const float*)d_in[5];
    const float* dWih  = (const float*)d_in[6];
    const float* dWhh  = (const float*)d_in[7];
    const float* dBih  = (const float*)d_in[8];
    const float* dBhh  = (const float*)d_in[9];
    const float* linW  = (const float*)d_in[10];
    const float* linB  = (const float*)d_in[11];
    const float* embed = (const float*)d_in[12];
    float* out = (float*)d_out;

    krnn_zero<<<(out_size + 255) / 256, 256, 0, stream>>>(out, out_size);

    dim3 grid((BN_TOT + ROWS_PER_BLOCK - 1) / ROWS_PER_BLOCK, NCLUST);
    krnn_gru<<<grid, 256, 0, stream>>>(X, eWih, eWhh, eBih, eBhh,
                                       dWih, dWhh, dBih, dBhh,
                                       linW, linB, embed, out);
}
